// Llama4TextExperts_1614907703546
// MI455X (gfx1250) — compile-verified, matrix operand order checked by reading, not run on hardware
//
#include <hip/hip_runtime.h>
#include <hip/hip_bf16.h>

// ---------------- problem constants ----------------
constexpr int E = 16;
constexpr int T = 2048;          // tokens per expert
constexpr int H = 1024;          // hidden
constexpr int I = 4096;          // intermediate
constexpr int TWO_I = 2 * I;

// ---------------- tiling ----------------
constexpr int BM = 128;          // token rows per block
constexpr int BN = 64;           // output cols per block (per matrix)
constexpr int BK = 64;           // K slab per LDS stage
// 256 threads = 8 waves, arranged 4 (M) x 2 (N); wave tile = 32x32 = 2x2 WMMA tiles

typedef __bf16 v16bf __attribute__((ext_vector_type(16)));
typedef float  v8f   __attribute__((ext_vector_type(8)));

union Frag { v16bf v; unsigned int u[8]; };

// fp32 -> bf16 (round to nearest even), raw u16 (scalar fallback path)
__device__ __forceinline__ unsigned short f2bf(float f) {
    unsigned int u = __builtin_bit_cast(unsigned int, f);
    u += 0x7FFFu + ((u >> 16) & 1u);
    return (unsigned short)(u >> 16);
}

// pack two floats (consecutive K) into one dword of bf16 pair
#if __has_builtin(__builtin_amdgcn_cvt_pk_bf16_f32)
typedef __bf16 v2bf __attribute__((ext_vector_type(2)));
__device__ __forceinline__ unsigned int pack2(float lo, float hi) {
    v2bf v = __builtin_amdgcn_cvt_pk_bf16_f32(lo, hi);
    return __builtin_bit_cast(unsigned int, v);
}
#else
__device__ __forceinline__ unsigned int pack2(float lo, float hi) {
    return (unsigned int)f2bf(lo) | ((unsigned int)f2bf(hi) << 16);
}
#endif

// LDS byte offset of a __shared__ object: flat-shared addr low 32 bits (ISA 10.2)
__device__ __forceinline__ unsigned int lds_off(const void* p) {
    return (unsigned int)(unsigned long long)p;
}

// Load a 16x32 bf16 WMMA fragment from an LDS tile stored row-major with
// K contiguous (stride = BK u16). CDNA5 16-bit A/B layout: lanes 0-15 = row,
// lane>=16 shifts K by +8 (dwords 0-3) / +24 (dwords 4-7).
__device__ __forceinline__ v16bf load_frag(const unsigned short* __restrict__ base,
                                           int row0, int k0, int lane) {
    const int r = row0 + (lane & 15);
    const int h = (lane >> 4) << 3;                 // 0 or 8
    const unsigned short* p = base + r * BK + k0 + h;
    Frag f;
#pragma unroll
    for (int v = 0; v < 4; ++v)
        f.u[v] = *(const unsigned int*)(p + 2 * v);
#pragma unroll
    for (int v = 0; v < 4; ++v)
        f.u[4 + v] = *(const unsigned int*)(p + 16 + 2 * v);
    return f.v;
}

// =====================================================================
// Kernel 1: gate_up GEMM + SwiGLU fused, bf16 output to workspace
//   grid = (I/BN, T/BM, E)
// =====================================================================
__global__ __launch_bounds__(256, 2)
void moe_gateup_swiglu(const float* __restrict__ hidden,
                       const float* __restrict__ gate_up,
                       unsigned short* __restrict__ gated_ws) {
    __shared__ __align__(16) unsigned short sA [BM][BK];  // x tile, [m][k]
    __shared__ __align__(16) unsigned short sBg[BN][BK];  // gate W, [n][k] (transposed)
    __shared__ __align__(16) unsigned short sBu[BN][BK];  // up   W, [n][k] (transposed)

    const int tid  = threadIdx.x;
    const int lane = tid & 31;
    const int wid  = tid >> 5;
    const int wr   = wid & 3;        // wave row  (M): 0..3
    const int wc   = wid >> 2;       // wave col  (N): 0..1

    const int e  = blockIdx.z;
    const int m0 = blockIdx.y * BM;
    const int n0 = blockIdx.x * BN;

    const float* __restrict__ xa = hidden + ((size_t)e * T + m0) * H;
    const float* __restrict__ wg = gate_up + (size_t)e * H * TWO_I + n0;
    const float* __restrict__ wu = wg + I;

    v8f accg[2][2] = {};
    v8f accu[2][2] = {};

    for (int kb = 0; kb < H; kb += BK) {
        if (kb + BK < H) {   // speculative prefetch of next K slab
            __builtin_prefetch(xa + (size_t)tid * H + kb + BK, 0, 1);
            __builtin_prefetch(wg + (size_t)(kb + BK) * TWO_I + tid, 0, 1);
        }
        // ---- stage A: 128x64 fp32 -> bf16, K contiguous, dword-pair stores ----
#pragma unroll
        for (int i = 0; i < 8; ++i) {
            const int idx = i * 256 + tid;
            const int row = idx >> 4;
            const int c   = (idx & 15) * 4;
            const float4 va = *(const float4*)(xa + (size_t)row * H + kb + c);
            uint2 p;
            p.x = pack2(va.x, va.y);
            p.y = pack2(va.z, va.w);
            *(uint2*)&sA[row][c] = p;
        }
        // ---- stage B (gate & up): load 2 K-rows per thread so the transposed
        //      LDS write is a packed {k,k+1} dword per n ----
#pragma unroll
        for (int i = 0; i < 2; ++i) {
            const int idx = i * 256 + tid;            // 0..511
            const int k   = (idx >> 4) * 2;           // 0,2,...,62
            const int c   = (idx & 15) * 4;           // n base
            const size_t go0 = (size_t)(kb + k) * TWO_I + c;
            const size_t go1 = go0 + TWO_I;
            const float4 g0 = *(const float4*)(wg + go0);
            const float4 g1 = *(const float4*)(wg + go1);
            const float4 u0 = *(const float4*)(wu + go0);
            const float4 u1 = *(const float4*)(wu + go1);
            *(unsigned int*)&sBg[c + 0][k] = pack2(g0.x, g1.x);
            *(unsigned int*)&sBg[c + 1][k] = pack2(g0.y, g1.y);
            *(unsigned int*)&sBg[c + 2][k] = pack2(g0.z, g1.z);
            *(unsigned int*)&sBg[c + 3][k] = pack2(g0.w, g1.w);
            *(unsigned int*)&sBu[c + 0][k] = pack2(u0.x, u1.x);
            *(unsigned int*)&sBu[c + 1][k] = pack2(u0.y, u1.y);
            *(unsigned int*)&sBu[c + 2][k] = pack2(u0.z, u1.z);
            *(unsigned int*)&sBu[c + 3][k] = pack2(u0.w, u1.w);
        }
        __syncthreads();

#pragma unroll
        for (int kk = 0; kk < BK; kk += 32) {
            v16bf a[2], bg[2], bu[2];
#pragma unroll
            for (int mi = 0; mi < 2; ++mi)
                a[mi] = load_frag(&sA[0][0], wr * 32 + mi * 16, kk, lane);
#pragma unroll
            for (int ni = 0; ni < 2; ++ni) {
                bg[ni] = load_frag(&sBg[0][0], wc * 32 + ni * 16, kk, lane);
                bu[ni] = load_frag(&sBu[0][0], wc * 32 + ni * 16, kk, lane);
            }
#pragma unroll
            for (int mi = 0; mi < 2; ++mi)
#pragma unroll
                for (int ni = 0; ni < 2; ++ni) {
                    accg[mi][ni] = __builtin_amdgcn_wmma_f32_16x16x32_bf16(
                        false, a[mi], false, bg[ni], (short)0, accg[mi][ni], false, false);
                    accu[mi][ni] = __builtin_amdgcn_wmma_f32_16x16x32_bf16(
                        false, a[mi], false, bu[ni], (short)0, accu[mi][ni], false, false);
                }
        }
        __syncthreads();
    }

    // ---- SwiGLU epilogue, bf16 store to ws (E,T,I) ----
    const int hl = lane >> 4;            // 0/1 -> M += 8
    const int nl = lane & 15;
#pragma unroll
    for (int mi = 0; mi < 2; ++mi)
#pragma unroll
        for (int ni = 0; ni < 2; ++ni)
#pragma unroll
            for (int r = 0; r < 8; ++r) {
                const float g = accg[mi][ni][r];
                const float u = accu[mi][ni][r];
                // up * silu(gate); rcp is plenty for a bf16 result
                const float s = u * g * __builtin_amdgcn_rcpf(1.0f + __expf(-g));
                const int t = m0 + wr * 32 + mi * 16 + r + 8 * hl;
                const int n = n0 + wc * 32 + ni * 16 + nl;
                gated_ws[((size_t)e * T + t) * I + n] = f2bf(s);
            }
}

// =====================================================================
// Kernel 2: down projection: gated(bf16) @ down(fp32->bf16) -> out fp32
//   grid = (H/BN, T/BM, E)
//   A tile staged with GLOBAL_LOAD_ASYNC_TO_LDS_B128 (no VGPR round-trip).
// =====================================================================
__global__ __launch_bounds__(256, 2)
void moe_down(const unsigned short* __restrict__ gated_ws,
              const float* __restrict__ down,
              float* __restrict__ out) {
    __shared__ __align__(16) unsigned short sA[BM][BK];   // gated tile, [m][k]
    __shared__ __align__(16) unsigned short sB[BN][BK];   // down W, [n][k] (transposed)

    const int tid  = threadIdx.x;
    const int lane = tid & 31;
    const int wid  = tid >> 5;
    const int wr   = wid & 3;
    const int wc   = wid >> 2;

    const int e  = blockIdx.z;
    const int m0 = blockIdx.y * BM;
    const int n0 = blockIdx.x * BN;

    const unsigned short* __restrict__ aw = gated_ws + ((size_t)e * T + m0) * I;
    const float* __restrict__ wd = down + (size_t)e * I * H + n0;

    v8f acc[2][2] = {};

    for (int kb = 0; kb < I; kb += BK) {
        if (kb + BK < I) {
            __builtin_prefetch(aw + (size_t)tid * I + kb + BK, 0, 1);
            __builtin_prefetch(wd + (size_t)(kb + BK) * H + tid, 0, 1);
        }
        // ---- stage A: already bf16 -> async global->LDS 128-bit copies ----
#pragma unroll
        for (int i = 0; i < 4; ++i) {
            const int idx = i * 256 + tid;
            const int row = idx >> 3;
            const int c   = (idx & 7) * 8;             // 8 u16 = 16 bytes
            const unsigned short* g = aw + (size_t)row * I + kb + c;
            const unsigned int l = lds_off(&sA[row][c]);
            asm volatile("global_load_async_to_lds_b128 %0, %1, off"
                         :: "v"(l), "v"(g) : "memory");
        }
        // ---- stage B: 2 K-rows per thread, packed-pair transposed stores ----
#pragma unroll
        for (int i = 0; i < 2; ++i) {
            const int idx = i * 256 + tid;
            const int k   = (idx >> 4) * 2;
            const int c   = (idx & 15) * 4;
            const float4 b0 = *(const float4*)(wd + (size_t)(kb + k) * H + c);
            const float4 b1 = *(const float4*)(wd + (size_t)(kb + k + 1) * H + c);
            *(unsigned int*)&sB[c + 0][k] = pack2(b0.x, b1.x);
            *(unsigned int*)&sB[c + 1][k] = pack2(b0.y, b1.y);
            *(unsigned int*)&sB[c + 2][k] = pack2(b0.z, b1.z);
            *(unsigned int*)&sB[c + 3][k] = pack2(b0.w, b1.w);
        }
        asm volatile("s_wait_asynccnt 0" ::: "memory");
        __syncthreads();

#pragma unroll
        for (int kk = 0; kk < BK; kk += 32) {
            v16bf a[2], b[2];
#pragma unroll
            for (int mi = 0; mi < 2; ++mi)
                a[mi] = load_frag(&sA[0][0], wr * 32 + mi * 16, kk, lane);
#pragma unroll
            for (int ni = 0; ni < 2; ++ni)
                b[ni] = load_frag(&sB[0][0], wc * 32 + ni * 16, kk, lane);
#pragma unroll
            for (int mi = 0; mi < 2; ++mi)
#pragma unroll
                for (int ni = 0; ni < 2; ++ni)
                    acc[mi][ni] = __builtin_amdgcn_wmma_f32_16x16x32_bf16(
                        false, a[mi], false, b[ni], (short)0, acc[mi][ni], false, false);
        }
        __syncthreads();
    }

    // ---- fp32 epilogue to out (E,T,H) ----
    const int hl = lane >> 4;
    const int nl = lane & 15;
#pragma unroll
    for (int mi = 0; mi < 2; ++mi)
#pragma unroll
        for (int ni = 0; ni < 2; ++ni)
#pragma unroll
            for (int r = 0; r < 8; ++r) {
                const int t = m0 + wr * 32 + mi * 16 + r + 8 * hl;
                const int n = n0 + wc * 32 + ni * 16 + nl;
                out[((size_t)e * T + t) * H + n] = acc[mi][ni][r];
            }
}

// =====================================================================
extern "C" void kernel_launch(void* const* d_in, const int* in_sizes, int n_in,
                              void* d_out, int out_size, void* d_ws, size_t ws_size,
                              hipStream_t stream) {
    const float* hidden  = (const float*)d_in[0];   // (32768, 1024) fp32
    const float* gate_up = (const float*)d_in[1];   // (16, 1024, 8192) fp32
    const float* down    = (const float*)d_in[2];   // (16, 4096, 1024) fp32
    float* out = (float*)d_out;                     // (32768, 1024) fp32
    unsigned short* gated = (unsigned short*)d_ws;  // (16, 2048, 4096) bf16 = 256 MB

    dim3 blk(256);
    moe_gateup_swiglu<<<dim3(I / BN, T / BM, E), blk, 0, stream>>>(hidden, gate_up, gated);
    moe_down        <<<dim3(H / BN, T / BM, E), blk, 0, stream>>>(gated, down, out);
}
